// ProteinGraphEncoder_13932873909138
// MI455X (gfx1250) — matrix operand order repaired and checked by reading, not compiled, not run on hardware
//
#include <hip/hip_runtime.h>
#include <hip/hip_bf16.h>

// ---------------------------------------------------------------------------
// GIN + virtual-node protein graph encoder for gfx1250 (MI455X).
//  - GEMMs: v_wmma_f32_16x16x32_bf16, fp32 accumulate.
//  - Weights pre-converted to transposed bf16 -> B fragments = 2x b128 loads.
//  - GIN GEMMs: block-cooperative, weights + A panel staged in LDS via
//    async global->LDS copies (guarded by __has_builtin, safe fallback).
//  - Edge aggregation / BN / pooling in fp32 with global atomics.
// ---------------------------------------------------------------------------

typedef __bf16 bf16_t;
typedef __attribute__((ext_vector_type(16))) __bf16 v16bf;
typedef __attribute__((ext_vector_type(8)))  __bf16 v8bf;
typedef __attribute__((ext_vector_type(8)))  float  v8f;
typedef __attribute__((ext_vector_type(4)))  int    v4i;

#define GN   100000   // nodes
#define GE   1600000  // edges
#define GB   200      // graphs
#define GDIN 1280
#define GD   128
#define GL   5

#if defined(__HIP_DEVICE_COMPILE__)
#if __has_builtin(__builtin_amdgcn_global_load_async_to_lds_b128)
#define HAVE_ASYNC_COPY 1
#endif
#endif

static __device__ __forceinline__ v8f wmma_bf16(v16bf a, v16bf b, v8f c) {
  // (neg_a, A, neg_b, B, c_mod, C, reuse_a, reuse_b)
  return __builtin_amdgcn_wmma_f32_16x16x32_bf16(false, a, false, b,
                                                 (short)0, c, false, false);
}

// 16-byte global->LDS copy: async path if available, else via register.
static __device__ __forceinline__ void g2s_copy16(const void* g, void* s) {
#ifdef HAVE_ASYNC_COPY
  __builtin_amdgcn_global_load_async_to_lds_b128(
      (__attribute__((address_space(1))) v4i*)g,
      (__attribute__((address_space(3))) v4i*)s, 0, 0);
#else
  *(float4*)s = *(const float4*)g;
#endif
}

static __device__ __forceinline__ void g2s_wait() {
#ifdef HAVE_ASYNC_COPY
#if __has_builtin(__builtin_amdgcn_s_wait_asynccnt)
  __builtin_amdgcn_s_wait_asynccnt(0);
#else
  asm volatile("s_wait_asynccnt 0x0" ::: "memory");
#endif
#endif
}

// ----- WMMA fragment builders ----------------------------------------------
// A 16x32 bf16 layout (ISA 7.12.2): lanes 0-15 -> K={0..7,16..23},
// lanes 16-31 -> K={8..15,24..31}.  B 32x16: lane group = K half.

static __device__ __forceinline__ v16bf frag_pair(v8bf lo, v8bf hi) {
  v16bf r;
#pragma unroll
  for (int e = 0; e < 8; ++e) {
    r[e] = lo[e];
    r[e + 8] = hi[e];
  }
  return r;
}

// B fragment: column-major bf16 weights; col_k0 = &W^T[n*K + k0]
static __device__ __forceinline__ v16bf load_b_frag(const bf16_t* col_k0,
                                                    int kg) {
  const v8bf* p = (const v8bf*)(col_k0 + kg * 8);
  return frag_pair(p[0], p[2]);
}

// A fragment from row-major bf16; row_kbase = &A[m*ld + k0 + kg*8]
static __device__ __forceinline__ v16bf load_a_frag_bf16(
    const bf16_t* row_kbase) {
  const v8bf* p = (const v8bf*)row_kbase;
  return frag_pair(p[0], p[2]);  // k..k+7  and  k+16..k+23
}

// A fragment from row-major fp32 with convert; row_kbase = &A[m*ld+k0+kg*8]
static __device__ __forceinline__ v16bf cvt_a_frag_f32(const float* row_kbase) {
  const float4* p = (const float4*)row_kbase;
  const float4 a0 = p[0], a1 = p[1], a2 = p[4], a3 = p[5];
  v16bf r;
  r[0] = (bf16_t)a0.x;  r[1] = (bf16_t)a0.y;
  r[2] = (bf16_t)a0.z;  r[3] = (bf16_t)a0.w;
  r[4] = (bf16_t)a1.x;  r[5] = (bf16_t)a1.y;
  r[6] = (bf16_t)a1.z;  r[7] = (bf16_t)a1.w;
  r[8] = (bf16_t)a2.x;  r[9] = (bf16_t)a2.y;
  r[10] = (bf16_t)a2.z; r[11] = (bf16_t)a2.w;
  r[12] = (bf16_t)a3.x; r[13] = (bf16_t)a3.y;
  r[14] = (bf16_t)a3.z; r[15] = (bf16_t)a3.w;
  return r;
}

// A fragment = s*mul + v*vmul (fp32), converted to bf16
static __device__ __forceinline__ v16bf cvt_a_frag_axpy(const float* s_kbase,
                                                        const float* v_kbase,
                                                        float mul, float vmul) {
  const float4* ps = (const float4*)s_kbase;
  const float4* pv = (const float4*)v_kbase;
  const float4 s0 = ps[0], s1 = ps[1], s2 = ps[4], s3 = ps[5];
  const float4 v0 = pv[0], v1 = pv[1], v2 = pv[4], v3 = pv[5];
  v16bf r;
  r[0] = (bf16_t)(s0.x * mul + v0.x * vmul);
  r[1] = (bf16_t)(s0.y * mul + v0.y * vmul);
  r[2] = (bf16_t)(s0.z * mul + v0.z * vmul);
  r[3] = (bf16_t)(s0.w * mul + v0.w * vmul);
  r[4] = (bf16_t)(s1.x * mul + v1.x * vmul);
  r[5] = (bf16_t)(s1.y * mul + v1.y * vmul);
  r[6] = (bf16_t)(s1.z * mul + v1.z * vmul);
  r[7] = (bf16_t)(s1.w * mul + v1.w * vmul);
  r[8] = (bf16_t)(s2.x * mul + v2.x * vmul);
  r[9] = (bf16_t)(s2.y * mul + v2.y * vmul);
  r[10] = (bf16_t)(s2.z * mul + v2.z * vmul);
  r[11] = (bf16_t)(s2.w * mul + v2.w * vmul);
  r[12] = (bf16_t)(s3.x * mul + v3.x * vmul);
  r[13] = (bf16_t)(s3.y * mul + v3.y * vmul);
  r[14] = (bf16_t)(s3.z * mul + v3.z * vmul);
  r[15] = (bf16_t)(s3.w * mul + v3.w * vmul);
  return r;
}

// ---------------------------------------------------------------------------
// Weight convert+transpose: w (K x 128 fp32, row-major) -> wt (128 x K bf16)
// ---------------------------------------------------------------------------
__global__ void wt_kernel(const float* __restrict__ w, bf16_t* __restrict__ wt,
                          int K) {
  const long idx = (long)blockIdx.x * blockDim.x + threadIdx.x;
  if (idx >= (long)K * GD) return;
  const int k = (int)(idx >> 7);
  const int n = (int)idx & 127;
  wt[(long)n * K + k] = (bf16_t)w[idx];
}

// ---------------------------------------------------------------------------
// h = x @ proj_w + proj_b     (100000x1280 * 1280x128), wave-per-tile
// ---------------------------------------------------------------------------
__global__ __launch_bounds__(128) void proj_gemm_kernel(
    const float* __restrict__ x, const bf16_t* __restrict__ wT,
    const float* __restrict__ bias, float* __restrict__ h) {
  const int wave = threadIdx.x >> 5;
  const int lane = threadIdx.x & 31;
  const int tile = blockIdx.x * 4 + wave;   // 50000 tiles
  const int mt = tile >> 3;
  const int nt = tile & 7;
  const int m = mt * 16 + (lane & 15);
  const int n = nt * 16 + (lane & 15);
  const int kg = lane >> 4;

  const float* arow = x + (long)m * GDIN + kg * 8;
  const bf16_t* bcol = wT + (long)n * GDIN;

  v8f acc = {};
  for (int k0 = 0; k0 < GDIN; k0 += 32) {
    __builtin_prefetch(arow + k0 + 256, 0, 1);  // global_prefetch_b8
    const v16bf a = cvt_a_frag_f32(arow + k0);
    const v16bf b = load_b_frag(bcol + k0, kg);
    acc = wmma_bf16(a, b, acc);
  }
  const float bn_ = bias[n];
#pragma unroll
  for (int r = 0; r < 8; ++r) {
    const int row = mt * 16 + kg * 8 + r;
    h[(long)row * GD + n] = acc[r] + bn_;
  }
}

// ---------------------------------------------------------------------------
// agg[dst] += h[src]   (one thread per edge x 4-float chunk)
// ---------------------------------------------------------------------------
__global__ void edge_agg_kernel(const float* __restrict__ h,
                                const int* __restrict__ src,
                                const int* __restrict__ dst,
                                float* __restrict__ agg) {
  const long idx = (long)blockIdx.x * blockDim.x + threadIdx.x;
  const int e = (int)(idx >> 5);
  if (e >= GE) return;
  const int c = ((int)idx & 31) * 4;
  const int s = src[e];
  const int d = dst[e];
  const float4 v = *reinterpret_cast<const float4*>(h + (long)s * GD + c);
  float* p = agg + (long)d * GD + c;
  atomicAdd(p + 0, v.x);
  atomicAdd(p + 1, v.y);
  atomicAdd(p + 2, v.z);
  atomicAdd(p + 3, v.w);
}

// ---------------------------------------------------------------------------
// t = relu( ((1+eps)*h + agg) @ w1 + b1 )  -> bf16
// Block = 8 waves, one 16-row M panel; weights (32KB) staged in LDS (async),
// A panel combined into LDS while staging.
// ---------------------------------------------------------------------------
__global__ __launch_bounds__(256) void gin_mlp1_kernel(
    const float* __restrict__ h, const float* __restrict__ agg,
    const bf16_t* __restrict__ w1T, const float* __restrict__ b1,
    const float* __restrict__ eps, int layer, bf16_t* __restrict__ t) {
  __shared__ __attribute__((aligned(16))) bf16_t sW[GD * GD];   // 32 KB
  __shared__ __attribute__((aligned(16))) float sU[16 * GD];    // 8 KB
  const int tid = threadIdx.x;
  const int mt = blockIdx.x;  // 0..6249
  const float onepe = 1.0f + eps[layer];

#pragma unroll
  for (int i = 0; i < 8; ++i) {  // 8 * 256 * 16B = 32 KB
    const int c = (tid + i * 256) * 16;
    g2s_copy16((const char*)w1T + c, (char*)sW + c);
  }
#pragma unroll
  for (int i = 0; i < 2; ++i) {  // u = (1+eps)*h + agg, 16x128 fp32
    const int q = tid + i * 256;
    const long g = (long)mt * (16 * GD) + (long)q * 4;
    const float4 hv = *(const float4*)(h + g);
    const float4 av = *(const float4*)(agg + g);
    float4 u;
    u.x = onepe * hv.x + av.x;
    u.y = onepe * hv.y + av.y;
    u.z = onepe * hv.z + av.z;
    u.w = onepe * hv.w + av.w;
    *(float4*)(sU + q * 4) = u;
  }
  g2s_wait();
  __syncthreads();

  const int lane = tid & 31;
  const int nt = tid >> 5;  // 0..7
  const int n = nt * 16 + (lane & 15);
  const int kg = lane >> 4;
  const float* arow = sU + (lane & 15) * GD + kg * 8;
  const bf16_t* bcol = sW + n * GD;

  v8f acc = {};
#pragma unroll
  for (int k0 = 0; k0 < GD; k0 += 32) {
    const v16bf a = cvt_a_frag_f32(arow + k0);
    const v16bf b = load_b_frag(bcol + k0, kg);
    acc = wmma_bf16(a, b, acc);
  }
  const float bn_ = b1[n];
#pragma unroll
  for (int r = 0; r < 8; ++r) {
    const int row = mt * 16 + kg * 8 + r;
    t[(long)row * GD + n] = (bf16_t)fmaxf(acc[r] + bn_, 0.0f);
  }
}

// ---------------------------------------------------------------------------
// z = t @ w2 + b2   (A already bf16; both operands staged in LDS)
// ---------------------------------------------------------------------------
__global__ __launch_bounds__(256) void gin_mlp2_kernel(
    const bf16_t* __restrict__ t, const bf16_t* __restrict__ w2T,
    const float* __restrict__ b2, float* __restrict__ z) {
  __shared__ __attribute__((aligned(16))) bf16_t sW[GD * GD];   // 32 KB
  __shared__ __attribute__((aligned(16))) bf16_t sA[16 * GD];   // 4 KB
  const int tid = threadIdx.x;
  const int mt = blockIdx.x;

#pragma unroll
  for (int i = 0; i < 8; ++i) {
    const int c = (tid + i * 256) * 16;
    g2s_copy16((const char*)w2T + c, (char*)sW + c);
  }
  {
    const char* tp = (const char*)(t + (long)mt * (16 * GD));
    g2s_copy16(tp + tid * 16, (char*)sA + tid * 16);  // 4 KB
  }
  g2s_wait();
  __syncthreads();

  const int lane = tid & 31;
  const int nt = tid >> 5;
  const int n = nt * 16 + (lane & 15);
  const int kg = lane >> 4;
  const bf16_t* arow = sA + (lane & 15) * GD + kg * 8;
  const bf16_t* bcol = sW + n * GD;

  v8f acc = {};
#pragma unroll
  for (int k0 = 0; k0 < GD; k0 += 32) {
    const v16bf a = load_a_frag_bf16(arow + k0);
    const v16bf b = load_b_frag(bcol + k0, kg);
    acc = wmma_bf16(a, b, acc);
  }
  const float bn_ = b2[n];
#pragma unroll
  for (int r = 0; r < 8; ++r) {
    const int row = mt * 16 + kg * 8 + r;
    z[(long)row * GD + n] = acc[r] + bn_;
  }
}

// ---------------------------------------------------------------------------
// BatchNorm stats / finalize / apply(+residual)
// ---------------------------------------------------------------------------
__global__ void bn_stats_kernel(const float* __restrict__ z,
                                float* __restrict__ sums) {
  __shared__ float s1[256];
  __shared__ float s2[256];
  const int col = threadIdx.x & 127;
  const int half = threadIdx.x >> 7;
  const int r0 = blockIdx.x * 400 + half;
  const int r1 = blockIdx.x * 400 + 400;
  float s = 0.0f, sq = 0.0f;
  for (int r = r0; r < r1; r += 2) {
    const float v = z[(long)r * GD + col];
    s += v;
    sq += v * v;
  }
  s1[threadIdx.x] = s;
  s2[threadIdx.x] = sq;
  __syncthreads();
  if (half == 0) {
    atomicAdd(&sums[col], s1[col] + s1[col + 128]);
    atomicAdd(&sums[GD + col], s2[col] + s2[col + 128]);
  }
}

__global__ void bn_finalize_kernel(const float* __restrict__ sums,
                                   const float* __restrict__ gamma,
                                   const float* __restrict__ beta,
                                   float* __restrict__ ss) {
  const int c = threadIdx.x;  // 128 threads
  const float inv = 1.0f / (float)GN;
  const float mean = sums[c] * inv;
  const float var = sums[GD + c] * inv - mean * mean;
  const float sc = gamma[c] * rsqrtf(var + 1e-5f);
  ss[c] = sc;
  ss[GD + c] = beta[c] - sc * mean;
}

__global__ void bn_apply_kernel(const float* __restrict__ z,
                                const float* __restrict__ ss,
                                float* __restrict__ h) {
  const long idx = (long)blockIdx.x * blockDim.x + threadIdx.x;
  if (idx >= (long)GN * GD) return;
  const int c = (int)(idx & (GD - 1));
  h[idx] += fmaxf(z[idx] * ss[c] + ss[GD + c], 0.0f);
}

// ---------------------------------------------------------------------------
// Per-graph pooling
// ---------------------------------------------------------------------------
__global__ void pool_kernel(const float* __restrict__ h,
                            const int* __restrict__ batch,
                            float* __restrict__ sums,
                            float* __restrict__ counts) {
  const long idx = (long)blockIdx.x * blockDim.x + threadIdx.x;
  const int nId = (int)(idx >> 5);
  if (nId >= GN) return;
  const int c = ((int)idx & 31) * 4;
  const int b = batch[nId];
  const float4 v = *reinterpret_cast<const float4*>(h + (long)nId * GD + c);
  float* p = sums + (long)b * GD + c;
  atomicAdd(p + 0, v.x);
  atomicAdd(p + 1, v.y);
  atomicAdd(p + 2, v.z);
  atomicAdd(p + 3, v.w);
  if (c == 0) atomicAdd(&counts[b], 1.0f);
}

// ---------------------------------------------------------------------------
// out = (sums/max(cnt,1) + vn*(cnt>0)) @ fc_w + fc_b    (200x128x128)
// ---------------------------------------------------------------------------
__global__ __launch_bounds__(128) void fc_kernel(
    const float* __restrict__ sums, const float* __restrict__ counts,
    const float* __restrict__ vn, const bf16_t* __restrict__ wT,
    const float* __restrict__ bias, float* __restrict__ out) {
  const int wave = threadIdx.x >> 5;
  const int lane = threadIdx.x & 31;
  const int tile = blockIdx.x * 4 + wave;  // 104 tiles
  const int mt = tile >> 3;
  const int nt = tile & 7;
  const int m = mt * 16 + (lane & 15);
  const int n = nt * 16 + (lane & 15);
  const int kg = lane >> 4;

  const bool mok = (m < GB);
  const int msafe = mok ? m : 0;
  const float cnt = counts[msafe];
  const float mul = mok ? (1.0f / fmaxf(cnt, 1.0f)) : 0.0f;
  const float vmul = (mok && cnt > 0.0f) ? 1.0f : 0.0f;

  const float* arow = sums + (long)msafe * GD + kg * 8;
  const float* vrow = vn + kg * 8;
  const bf16_t* bcol = wT + (long)n * GD;

  v8f acc = {};
#pragma unroll
  for (int k0 = 0; k0 < GD; k0 += 32) {
    const v16bf a = cvt_a_frag_axpy(arow + k0, vrow + k0, mul, vmul);
    const v16bf b = load_b_frag(bcol + k0, kg);
    acc = wmma_bf16(a, b, acc);
  }
  const float bn_ = bias[n];
#pragma unroll
  for (int r = 0; r < 8; ++r) {
    const int row = mt * 16 + kg * 8 + r;
    if (row < GB) out[(long)row * GD + n] = acc[r] + bn_;
  }
}

// ---------------------------------------------------------------------------
extern "C" void kernel_launch(void* const* d_in, const int* in_sizes, int n_in,
                              void* d_out, int out_size, void* d_ws,
                              size_t ws_size, hipStream_t stream) {
  (void)in_sizes; (void)n_in; (void)out_size; (void)ws_size;

  const float* x      = (const float*)d_in[0];
  const int*   ei     = (const int*)d_in[1];
  const int*   batch  = (const int*)d_in[2];
  const float* proj_w = (const float*)d_in[3];
  const float* proj_b = (const float*)d_in[4];
  const float* gw1    = (const float*)d_in[5];
  const float* gb1    = (const float*)d_in[6];
  const float* gw2    = (const float*)d_in[7];
  const float* gb2    = (const float*)d_in[8];
  const float* eps    = (const float*)d_in[9];
  const float* gamma  = (const float*)d_in[10];
  const float* beta   = (const float*)d_in[11];
  const float* vn     = (const float*)d_in[12];
  const float* fcw    = (const float*)d_in[13];
  const float* fcb    = (const float*)d_in[14];
  float* out = (float*)d_out;

  const long ND = (long)GN * GD;  // 12.8M
  float* ws = (float*)d_ws;
  float*  h    = ws;                      // ND floats
  float*  agg  = ws + ND;                 // ND floats (z aliases after mlp1)
  float*  z    = agg;
  bf16_t* tb   = (bf16_t*)(ws + 2 * ND);  // ND bf16 (= ND/2 float slots)
  float*  p3   = ws + 2 * ND + ND / 2;
  bf16_t* wprojT = (bf16_t*)p3;                        // 1280*128 bf16
  bf16_t* g1T    = (bf16_t*)(p3 + 81920);              // 5*128*128 bf16
  bf16_t* g2T    = (bf16_t*)(p3 + 81920 + 40960);
  bf16_t* fcT    = (bf16_t*)(p3 + 81920 + 2 * 40960);  // 128*128 bf16
  float*  bns  = p3 + 81920 + 2 * 40960 + 8192;        // 2*D
  float*  ss   = bns + 2 * GD;                         // 2*D
  float*  psum = ss + 2 * GD;                          // B*D
  float*  pcnt = psum + (long)GB * GD;                 // B

  const int* src = ei;
  const int* dst = ei + GE;

  // --- weight prep (bf16 transpose) ---
  wt_kernel<<<(GDIN * GD + 255) / 256, 256, 0, stream>>>(proj_w, wprojT, GDIN);
  for (int l = 0; l < GL; ++l) {
    wt_kernel<<<64, 256, 0, stream>>>(gw1 + (long)l * GD * GD,
                                      g1T + (long)l * GD * GD, GD);
    wt_kernel<<<64, 256, 0, stream>>>(gw2 + (long)l * GD * GD,
                                      g2T + (long)l * GD * GD, GD);
  }
  wt_kernel<<<64, 256, 0, stream>>>(fcw, fcT, GD);

  // --- projection ---
  proj_gemm_kernel<<<12500, 128, 0, stream>>>(x, wprojT, proj_b, h);

  const int eGrid = (int)(((long)GE * 32 + 255) / 256);
  const int ndGrid = (int)((ND + 255) / 256);
  const int poolGrid = (int)(((long)GN * 32 + 255) / 256);
  const int ginGrid = GN / 16;  // 6250

  for (int i = 0; i < GL; ++i) {
    (void)hipMemsetAsync(agg, 0, ND * sizeof(float), stream);
    edge_agg_kernel<<<eGrid, 256, 0, stream>>>(h, src, dst, agg);
    gin_mlp1_kernel<<<ginGrid, 256, 0, stream>>>(
        h, agg, g1T + (long)i * GD * GD, gb1 + i * GD, eps, i, tb);
    gin_mlp2_kernel<<<ginGrid, 256, 0, stream>>>(
        tb, g2T + (long)i * GD * GD, gb2 + i * GD, z);
    (void)hipMemsetAsync(bns, 0, 2 * GD * sizeof(float), stream);
    bn_stats_kernel<<<250, 256, 0, stream>>>(z, bns);
    bn_finalize_kernel<<<1, GD, 0, stream>>>(bns, gamma + i * GD,
                                             beta + i * GD, ss);
    bn_apply_kernel<<<ndGrid, 256, 0, stream>>>(z, ss, h);
  }

  (void)hipMemsetAsync(psum, 0, ((long)GB * GD + GB) * sizeof(float), stream);
  pool_kernel<<<poolGrid, 256, 0, stream>>>(h, batch, psum, pcnt);
  fc_kernel<<<26, 128, 0, stream>>>(psum, pcnt, vn, fcT, fcb, out);
}
